// Model_11106785428041
// MI455X (gfx1250) — matrix operand order repaired and checked by reading, not compiled, not run on hardware
//
#include <hip/hip_runtime.h>
#include <hip/hip_bf16.h>
#include <math.h>

// Problem dims (fixed by reference)
#define LREP 5
#define BB   4
#define NN   16384
#define HH   8
#define DHH  32
#define GG   32
#define HIDN 256
#define OUTC 4
#define MTOK (BB * NN)          // 65536 tokens
#define NCHUNK 32               // deterministic partial-sum chunks over N

typedef __bf16 v16bf __attribute__((ext_vector_type(16)));
typedef float  v8f   __attribute__((ext_vector_type(8)));

__device__ __forceinline__ unsigned short f32_to_bf16(float f) {
    unsigned int u = __float_as_uint(f);
    u += 0x7FFFu + ((u >> 16) & 1u);          // round-to-nearest-even
    return (unsigned short)(u >> 16);
}
__device__ __forceinline__ float bf16_to_f32(unsigned short h) {
    return __uint_as_float(((unsigned int)h) << 16);
}
__device__ __forceinline__ float gelu_f(float v) {
    return 0.5f * v * (1.f + erff(v * 0.70710678118654752f));
}

// ---------------------------------------------------------------------------
// Weight conversion f32 [K][256] -> bf16 TRANSPOSED [256][K]
// (reads coalesced; scattered writes; runs once per launch, tiny)
// ---------------------------------------------------------------------------
__global__ void k_conv_w_t(const float* __restrict__ w, unsigned short* __restrict__ wt, int Kdim) {
    int total = Kdim * HIDN;
    for (int idx = blockIdx.x * 256 + threadIdx.x; idx < total; idx += gridDim.x * 256) {
        int k = idx / HIDN, n = idx % HIDN;
        wt[(size_t)n * Kdim + k] = f32_to_bf16(w[idx]);
    }
}

// ---------------------------------------------------------------------------
// Preprocess stage 1: hid = gelu(x @ pre_w1 + pre_b1)  (K=2, VALU)
// ---------------------------------------------------------------------------
__global__ void k_pre1(const float* __restrict__ x, const float* __restrict__ w1,
                       const float* __restrict__ b1, unsigned short* __restrict__ hid) {
    size_t idx = (size_t)blockIdx.x * 256 + threadIdx.x;   // M*512 threads
    int j = (int)(idx & 511);
    size_t t = idx >> 9;
    float v = b1[j] + x[t * 2] * w1[j] + x[t * 2 + 1] * w1[512 + j];
    hid[idx] = f32_to_bf16(gelu_f(v));
}

// fx += placeholder + (condition @ emb_w + emb_b)
__global__ void k_add_pc(float* __restrict__ fx, const float* __restrict__ ph,
                         const float* __restrict__ cond, const float* __restrict__ ew,
                         const float* __restrict__ eb) {
    size_t idx = (size_t)blockIdx.x * 256 + threadIdx.x;   // M*256
    int j = (int)(idx & 255);
    size_t t = idx >> 8;
    int b = (int)(t / NN);
    float ce = eb[j];
    for (int k = 0; k < 3; ++k) ce += cond[b * 3 + k] * ew[k * HIDN + j];
    fx[idx] += ph[j] + ce;
}

// ---------------------------------------------------------------------------
// LayerNorm over HID=256, one wave per token, bf16 out
// ---------------------------------------------------------------------------
__global__ __launch_bounds__(256) void k_ln(const float* __restrict__ x,
                                            const float* __restrict__ g,
                                            const float* __restrict__ b,
                                            unsigned short* __restrict__ out) {
    int wave = threadIdx.x >> 5, lane = threadIdx.x & 31;
    size_t t = (size_t)blockIdx.x * 8 + wave;
    const float* row = x + t * HIDN;
    float vals[8], s = 0.f;
#pragma unroll
    for (int i = 0; i < 8; ++i) { vals[i] = row[lane + 32 * i]; s += vals[i]; }
    for (int off = 16; off > 0; off >>= 1) s += __shfl_xor(s, off, 32);
    float mean = s * (1.f / 256.f);
    float vs = 0.f;
#pragma unroll
    for (int i = 0; i < 8; ++i) { float d = vals[i] - mean; vs += d * d; }
    for (int off = 16; off > 0; off >>= 1) vs += __shfl_xor(vs, off, 32);
    float inv = rsqrtf(vs * (1.f / 256.f) + 1e-5f);
#pragma unroll
    for (int i = 0; i < 8; ++i) {
        int j = lane + 32 * i;
        out[t * HIDN + j] = f32_to_bf16((vals[i] - mean) * inv * g[j] + b[j]);
    }
}

// ---------------------------------------------------------------------------
// WMMA GEMM: out[M,256] = act(A_bf16[M,K] @ W + bias [+ resid]) with W given
// TRANSPOSED as Wt[256][K]. block = 128 (4 waves). Each block: 32 rows x 64 cols.
// Each wave: 32x16 tile = two accumulators sharing one B fragment.
// A tile (contiguous span) and Wt tile (contiguous span) staged as b128 copies.
// ---------------------------------------------------------------------------
template <int K, bool GELU, bool RESID, bool OUTF, bool OUTB>
__global__ __launch_bounds__(128) void k_gemm(const unsigned short* __restrict__ A,
                                              const unsigned short* __restrict__ Wt,
                                              const float* __restrict__ bias,
                                              float* __restrict__ fxio,
                                              unsigned short* __restrict__ outB) {
    __shared__ alignas(16) unsigned short lA[32 * K];
    __shared__ alignas(16) unsigned short lWt[64 * K];
    const int tid = threadIdx.x;
    const int m0 = blockIdx.x * 32;
    const int n0 = blockIdx.y * 64;
    __builtin_prefetch(&Wt[(size_t)n0 * K], 0, 3);   // global_prefetch_b8 on weight tile

    // A rows m0..m0+31 are one contiguous span; Wt rows n0..n0+63 likewise.
    const uint4* gA = (const uint4*)(A + (size_t)m0 * K);
    uint4* sA = (uint4*)lA;
    for (int idx = tid; idx < 32 * K / 8; idx += 128) sA[idx] = gA[idx];
    const uint4* gW = (const uint4*)(Wt + (size_t)n0 * K);
    uint4* sW = (uint4*)lWt;
    for (int idx = tid; idx < 64 * K / 8; idx += 128) sW[idx] = gW[idx];
    __syncthreads();

    const int lane = tid & 31, wave = tid >> 5;
    const bool lo = (lane < 16);
    const int mr = lane & 15;
    const unsigned int* rowA0 = (const unsigned int*)&lA[mr * K];
    const unsigned int* rowA1 = (const unsigned int*)&lA[(16 + mr) * K];
    const unsigned int* rowW  = (const unsigned int*)&lWt[(wave * 16 + mr) * K];
    v8f acc0 = {0.f, 0.f, 0.f, 0.f, 0.f, 0.f, 0.f, 0.f};
    v8f acc1 = {0.f, 0.f, 0.f, 0.f, 0.f, 0.f, 0.f, 0.f};

    for (int ks = 0; ks < K; ks += 32) {
        union { v16bf v; unsigned int u[8]; } af0, af1, bf;
        int a0 = (ks + (lo ? 0 : 8)) >> 1;        // dword index into A row
        int a1 = (ks + (lo ? 16 : 24)) >> 1;
#pragma unroll
        for (int j = 0; j < 4; ++j) {
            af0.u[j] = rowA0[a0 + j]; af0.u[4 + j] = rowA0[a1 + j];
            af1.u[j] = rowA1[a0 + j]; af1.u[4 + j] = rowA1[a1 + j];
        }
        int b0 = (ks + (lo ? 0 : 8)) >> 1;
        int b1 = (ks + (lo ? 16 : 24)) >> 1;
#pragma unroll
        for (int j = 0; j < 4; ++j) { bf.u[j] = rowW[b0 + j]; bf.u[4 + j] = rowW[b1 + j]; }
        acc0 = __builtin_amdgcn_wmma_f32_16x16x32_bf16(false, af0.v, false, bf.v,
                                                       (short)0, acc0, false, false);
        acc1 = __builtin_amdgcn_wmma_f32_16x16x32_bf16(false, af1.v, false, bf.v,
                                                       (short)0, acc1, false, false);
    }

    const int nc = n0 + wave * 16 + mr;
    const float bv = bias[nc];
#pragma unroll
    for (int j = 0; j < 8; ++j) {
        int mA = m0 + (lo ? j : j + 8);
        int mB = mA + 16;
        float v0 = acc0[j] + bv;
        float v1 = acc1[j] + bv;
        if (RESID) { v0 += fxio[(size_t)mA * HIDN + nc]; v1 += fxio[(size_t)mB * HIDN + nc]; }
        if (GELU)  { v0 = gelu_f(v0); v1 = gelu_f(v1); }
        if (OUTF)  { fxio[(size_t)mA * HIDN + nc] = v0; fxio[(size_t)mB * HIDN + nc] = v1; }
        if (OUTB)  { outB[(size_t)mA * HIDN + nc] = f32_to_bf16(v0);
                     outB[(size_t)mB * HIDN + nc] = f32_to_bf16(v1); }
    }
}

// ---------------------------------------------------------------------------
// Per (b,h,n): learned temperature + gumbel-softmax routing weights (VALU; tiny)
// ---------------------------------------------------------------------------
__global__ __launch_bounds__(256) void k_perhead(const unsigned short* __restrict__ xm_bf,
                                                 const float* __restrict__ gum,
                                                 const float* __restrict__ ptw1,
                                                 const float* __restrict__ ptb1,
                                                 const float* __restrict__ ptw2,
                                                 const float* __restrict__ ptb2,
                                                 const float* __restrict__ abias,
                                                 const float* __restrict__ psw,
                                                 const float* __restrict__ psb,
                                                 unsigned short* __restrict__ sw_bf, int l) {
    int idx = blockIdx.x * 256 + threadIdx.x;          // B*H*N threads
    int n = idx % NN, bh = idx / NN;
    int h = bh % HH, b = bh / HH;
    size_t t = (size_t)b * NN + n;
    float xv[32];
#pragma unroll
    for (int c = 0; c < 32; ++c) xv[c] = bf16_to_f32(xm_bf[t * HIDN + h * 32 + c]);

    const float* w1 = ptw1 + (size_t)l * 32 * 32;
    const float* s1 = psw + (size_t)l * 32 * 32;
    float t1[32];
#pragma unroll
    for (int gi = 0; gi < 32; ++gi) t1[gi] = ptb1[l * 32 + gi];
    for (int c = 0; c < 32; ++c) {
        float xc = xv[c];
#pragma unroll
        for (int gi = 0; gi < 32; ++gi) t1[gi] += xc * w1[c * 32 + gi];
    }
    float tv = ptb2[l];
#pragma unroll
    for (int gi = 0; gi < 32; ++gi) tv += gelu_f(t1[gi]) * ptw2[l * 32 + gi];
    float temp = fmaxf(gelu_f(tv) + abias[l * HH + h], 0.01f);

    const float* gu = gum + ((((size_t)l * BB + b) * HH + h) * NN + n) * GG;
    float sv[32], mx = -1e30f;
    for (int gi = 0; gi < 32; ++gi) {
        float s = psb[l * 32 + gi];
#pragma unroll
        for (int c = 0; c < 32; ++c) s += xv[c] * s1[c * 32 + gi];
        float gn = -logf(-logf(gu[gi] + 1e-8f) + 1e-8f);
        s = (s + gn) / temp;
        sv[gi] = s; mx = fmaxf(mx, s);
    }
    float den = 0.f;
#pragma unroll
    for (int gi = 0; gi < 32; ++gi) { sv[gi] = expf(sv[gi] - mx); den += sv[gi]; }
    float r = 1.f / den;
    size_t base = ((size_t)bh * NN + n) * GG;
#pragma unroll
    for (int gi = 0; gi < 32; ++gi) sw_bf[base + gi] = f32_to_bf16(sv[gi] * r);
}

// ---------------------------------------------------------------------------
// st partial: st[g,c] = sum_n sw[n,g]*xm[n,c] over a 512-token chunk (WMMA K-red.)
// grid (B*H, NCHUNK), block 128 (4 waves -> 2x2 quadrants of the 32x32 output).
// sw and xm are staged TRANSPOSED into LDS so fragments build from contiguous
// dwords: lswT[g][n] feeds A, lxmT[c][n] feeds B.
// ---------------------------------------------------------------------------
__global__ __launch_bounds__(128) void k_slice_part(const unsigned short* __restrict__ sw_bf,
                                                    const unsigned short* __restrict__ xm_bf,
                                                    float* __restrict__ st_part,
                                                    float* __restrict__ norm_part) {
    __shared__ alignas(16) unsigned short lswT[32 * 32];   // [g][n]
    __shared__ alignas(16) unsigned short lxmT[32 * 32];   // [c][n]
    int bh = blockIdx.x, chunk = blockIdx.y;
    int b = bh / HH, h = bh % HH;
    int tid = threadIdx.x, lane = tid & 31, wave = tid >> 5;
    int gh = wave >> 1, ch = wave & 1;
    const bool lo = (lane < 16);
    const int mrow = (lane & 15) + gh * 16;
    const int col = (lane & 15) + ch * 16;
    v8f acc = {0.f, 0.f, 0.f, 0.f, 0.f, 0.f, 0.f, 0.f};
    float nacc = 0.f;
    int nbase0 = chunk * (NN / NCHUNK);

    for (int nb = 0; nb < NN / NCHUNK; nb += 32) {
        for (int idx = tid; idx < 1024; idx += 128) {
            int nn = idx >> 5, c = idx & 31;        // coalesced global reads over c
            int n = nbase0 + nb + nn;
            lswT[c * 32 + nn] = sw_bf[((size_t)bh * NN + n) * GG + c];
            lxmT[c * 32 + nn] = xm_bf[((size_t)b * NN + n) * HIDN + h * 32 + c];
        }
        __syncthreads();
        if (tid < 32) {   // row of lswT is contiguous now
            for (int nn = 0; nn < 32; ++nn) nacc += bf16_to_f32(lswT[tid * 32 + nn]);
        }
        union { v16bf v; unsigned int u[8]; } af, bfr;
        const unsigned int* rA = (const unsigned int*)&lswT[mrow * 32];
        const unsigned int* rB = (const unsigned int*)&lxmT[col * 32];
        int a0 = lo ? 0 : 4, a1 = lo ? 8 : 12;
#pragma unroll
        for (int j = 0; j < 4; ++j) { af.u[j] = rA[a0 + j]; af.u[4 + j] = rA[a1 + j]; }
        int b0 = lo ? 0 : 4, b1 = lo ? 8 : 12;
#pragma unroll
        for (int j = 0; j < 4; ++j) { bfr.u[j] = rB[b0 + j]; bfr.u[4 + j] = rB[b1 + j]; }
        acc = __builtin_amdgcn_wmma_f32_16x16x32_bf16(false, af.v, false, bfr.v,
                                                      (short)0, acc, false, false);
        __syncthreads();
    }
#pragma unroll
    for (int j = 0; j < 8; ++j) {
        int g = (lo ? j : j + 8) + gh * 16;
        st_part[(((size_t)chunk * 32 + bh) * GG + g) * DHH + col] = acc[j];
    }
    if (tid < 32) norm_part[((size_t)chunk * 32 + bh) * GG + tid] = nacc;
}

// ---------------------------------------------------------------------------
// Attention over 32 slice tokens per (b,h). Writes o TRANSPOSED: oT[c][g].
// ---------------------------------------------------------------------------
__global__ __launch_bounds__(32) void k_attn(const float* __restrict__ st_part,
                                             const float* __restrict__ norm_part,
                                             const float* __restrict__ qw,
                                             const float* __restrict__ kw,
                                             const float* __restrict__ vw,
                                             unsigned short* __restrict__ oT_bf, int l) {
    __shared__ float stn[32][32];
    __shared__ float kk[32][32];
    __shared__ float vv[32][32];
    int bh = blockIdx.x, g = threadIdx.x;
    float norm = 0.f;
    for (int p = 0; p < NCHUNK; ++p) norm += norm_part[((size_t)p * 32 + bh) * GG + g];
    float inv = 1.f / (norm + 1e-5f);
    for (int c = 0; c < 32; ++c) {
        float s = 0.f;
        for (int p = 0; p < NCHUNK; ++p) s += st_part[(((size_t)p * 32 + bh) * GG + g) * DHH + c];
        stn[g][c] = s * inv;
    }
    __syncthreads();
    const float* Q = qw + (size_t)l * 32 * 32;
    const float* Kw = kw + (size_t)l * 32 * 32;
    const float* Vw = vw + (size_t)l * 32 * 32;
    float qr[32];
    for (int c2 = 0; c2 < 32; ++c2) {
        float q = 0.f, k = 0.f, v = 0.f;
        for (int c = 0; c < 32; ++c) {
            float x = stn[g][c];
            q += x * Q[c * 32 + c2]; k += x * Kw[c * 32 + c2]; v += x * Vw[c * 32 + c2];
        }
        qr[c2] = q; kk[g][c2] = k; vv[g][c2] = v;
    }
    __syncthreads();
    const float scale = 0.1767766952966369f;   // 1/sqrt(32)
    float sc[32], mx = -1e30f;
    for (int k2 = 0; k2 < 32; ++k2) {
        float s = 0.f;
        for (int c = 0; c < 32; ++c) s += qr[c] * kk[k2][c];
        sc[k2] = s * scale; mx = fmaxf(mx, sc[k2]);
    }
    float den = 0.f;
    for (int k2 = 0; k2 < 32; ++k2) { sc[k2] = expf(sc[k2] - mx); den += sc[k2]; }
    float r = 1.f / den;
    for (int c = 0; c < 32; ++c) {
        float o = 0.f;
        for (int k2 = 0; k2 < 32; ++k2) o += sc[k2] * vv[k2][c];
        oT_bf[((size_t)bh * DHH + c) * GG + g] = f32_to_bf16(o * r);   // transposed
    }
}

// ---------------------------------------------------------------------------
// Scatter-back: ox[n, h*32+c] = sum_g sw[n,g]*oT[c,g]  (K=32 -> one WMMA step)
// grid (B*H, N/64), block 128; A rows and B rows both contiguous dwords.
// ---------------------------------------------------------------------------
__global__ __launch_bounds__(128) void k_scatter(const unsigned short* __restrict__ sw_bf,
                                                 const unsigned short* __restrict__ oT_bf,
                                                 unsigned short* __restrict__ ox_bf) {
    int bh = blockIdx.x, b = bh / HH, h = bh % HH;
    int lane = threadIdx.x & 31, wave = threadIdx.x >> 5;
    const bool lo = (lane < 16);
    int n0 = blockIdx.y * 64 + wave * 16;
    int tr = n0 + (lane & 15);
    const unsigned int* rowA = (const unsigned int*)(sw_bf + ((size_t)bh * NN + tr) * GG);
    union { v16bf v; unsigned int u[8]; } af;
    int a0 = lo ? 0 : 4, a1 = lo ? 8 : 12;
#pragma unroll
    for (int j = 0; j < 4; ++j) { af.u[j] = rowA[a0 + j]; af.u[4 + j] = rowA[a1 + j]; }
    int b0 = lo ? 0 : 4, b1 = lo ? 8 : 12;
#pragma unroll
    for (int ct = 0; ct < 2; ++ct) {
        int col = (lane & 15) + ct * 16;
        const unsigned int* rowB = (const unsigned int*)(oT_bf + ((size_t)bh * DHH + col) * GG);
        union { v16bf v; unsigned int u[8]; } bfr;
#pragma unroll
        for (int j = 0; j < 4; ++j) { bfr.u[j] = rowB[b0 + j]; bfr.u[4 + j] = rowB[b1 + j]; }
        v8f acc = {0.f, 0.f, 0.f, 0.f, 0.f, 0.f, 0.f, 0.f};
        acc = __builtin_amdgcn_wmma_f32_16x16x32_bf16(false, af.v, false, bfr.v,
                                                      (short)0, acc, false, false);
#pragma unroll
        for (int j = 0; j < 8; ++j) {
            int m = lo ? j : j + 8;
            size_t t = (size_t)b * NN + n0 + m;
            ox_bf[t * HIDN + h * 32 + col] = f32_to_bf16(acc[j]);
        }
    }
}

// ---------------------------------------------------------------------------
// Head: out[t, oc] = ln3(fx)[t] @ head_w + head_b  (256 -> 4, VALU)
// ---------------------------------------------------------------------------
__global__ __launch_bounds__(256) void k_head(const unsigned short* __restrict__ hbf,
                                              const float* __restrict__ hw,
                                              const float* __restrict__ hb,
                                              float* __restrict__ out) {
    int idx = blockIdx.x * 256 + threadIdx.x;   // M*4 threads
    int t = idx >> 2, oc = idx & 3;
    float s = hb[oc];
    const unsigned short* row = hbf + (size_t)t * HIDN;
    for (int c = 0; c < HIDN; ++c) s += bf16_to_f32(row[c]) * hw[c * 4 + oc];
    out[idx] = s;
}

// ---------------------------------------------------------------------------
extern "C" void kernel_launch(void* const* d_in, const int* in_sizes, int n_in,
                              void* d_out, int out_size, void* d_ws, size_t ws_size,
                              hipStream_t stream) {
    (void)in_sizes; (void)n_in; (void)out_size; (void)ws_size;
    const float* x         = (const float*)d_in[0];
    const float* condition = (const float*)d_in[1];
    const float* gumbel    = (const float*)d_in[2];
    const float* pre_w1    = (const float*)d_in[3];
    const float* pre_b1    = (const float*)d_in[4];
    const float* pre_w2    = (const float*)d_in[5];
    const float* pre_b2    = (const float*)d_in[6];
    const float* placeh    = (const float*)d_in[7];
    const float* emb_w     = (const float*)d_in[8];
    const float* emb_b     = (const float*)d_in[9];
    const float* ln1_g     = (const float*)d_in[10];
    const float* ln1_b     = (const float*)d_in[11];
    const float* attn_bias = (const float*)d_in[12];
    const float* pt_w1     = (const float*)d_in[13];
    const float* pt_b1     = (const float*)d_in[14];
    const float* pt_w2     = (const float*)d_in[15];
    const float* pt_b2     = (const float*)d_in[16];
    const float* px_w      = (const float*)d_in[17];
    const float* px_b      = (const float*)d_in[18];
    const float* ps_w      = (const float*)d_in[19];
    const float* ps_b      = (const float*)d_in[20];
    const float* q_w       = (const float*)d_in[21];
    const float* k_w       = (const float*)d_in[22];
    const float* v_w       = (const float*)d_in[23];
    const float* o_w       = (const float*)d_in[24];
    const float* o_b       = (const float*)d_in[25];
    const float* ln2_g     = (const float*)d_in[26];
    const float* ln2_b     = (const float*)d_in[27];
    const float* mlp_w1    = (const float*)d_in[28];
    const float* mlp_b1    = (const float*)d_in[29];
    const float* mlp_w2    = (const float*)d_in[30];
    const float* mlp_b2    = (const float*)d_in[31];
    const float* ln3_g     = (const float*)d_in[32];
    const float* ln3_b     = (const float*)d_in[33];
    const float* head_w    = (const float*)d_in[34];
    const float* head_b    = (const float*)d_in[35];

    // Workspace carving (≈260 MB)
    char* ws = (char*)d_ws;
    size_t off = 0;
    auto carve = [&](size_t bytes) -> char* {
        char* p = ws + off;
        off = (off + bytes + 255) & ~(size_t)255;
        return p;
    };
    float*          fx       = (float*)         carve((size_t)MTOK * HIDN * 4);
    unsigned short* hid_bf   = (unsigned short*)carve((size_t)MTOK * 512 * 2);
    unsigned short* h_bf     = (unsigned short*)carve((size_t)MTOK * HIDN * 2);
    unsigned short* xm_bf    = (unsigned short*)carve((size_t)MTOK * HIDN * 2);
    unsigned short* sw_bf    = (unsigned short*)carve((size_t)BB * HH * NN * GG * 2);
    unsigned short* mid_bf   = (unsigned short*)carve((size_t)MTOK * HIDN * 2);  // ox, then mlp mid
    float*          st_part  = (float*)         carve((size_t)NCHUNK * 32 * GG * DHH * 4);
    float*          norm_prt = (float*)         carve((size_t)NCHUNK * 32 * GG * 4);
    unsigned short* oT_bf    = (unsigned short*)carve((size_t)BB * HH * GG * DHH * 2);
    unsigned short* wpre2_bf = (unsigned short*)carve((size_t)512 * HIDN * 2);
    unsigned short* wpx_bf   = (unsigned short*)carve((size_t)LREP * HIDN * HIDN * 2);
    unsigned short* wo_bf    = (unsigned short*)carve((size_t)LREP * HIDN * HIDN * 2);
    unsigned short* wm1_bf   = (unsigned short*)carve((size_t)LREP * HIDN * HIDN * 2);
    unsigned short* wm2_bf   = (unsigned short*)carve((size_t)LREP * HIDN * HIDN * 2);

    // Weight conversions (transposed bf16), deterministic, every launch
    k_conv_w_t<<<256, 256, 0, stream>>>(pre_w2, wpre2_bf, 512);
    for (int l = 0; l < LREP; ++l) {
        size_t wofs = (size_t)l * HIDN * HIDN;
        k_conv_w_t<<<128, 256, 0, stream>>>(px_w   + wofs, wpx_bf + wofs, HIDN);
        k_conv_w_t<<<128, 256, 0, stream>>>(o_w    + wofs, wo_bf  + wofs, HIDN);
        k_conv_w_t<<<128, 256, 0, stream>>>(mlp_w1 + wofs, wm1_bf + wofs, HIDN);
        k_conv_w_t<<<128, 256, 0, stream>>>(mlp_w2 + wofs, wm2_bf + wofs, HIDN);
    }

    const dim3 gGemm(MTOK / 32, HIDN / 64);   // (2048, 4)

    // Preprocess
    k_pre1<<<(MTOK * 512) / 256, 256, 0, stream>>>(x, pre_w1, pre_b1, hid_bf);
    k_gemm<512, false, false, true, false><<<gGemm, 128, 0, stream>>>(
        hid_bf, wpre2_bf, pre_b2, fx, h_bf);
    k_add_pc<<<(MTOK * HIDN) / 256, 256, 0, stream>>>(fx, placeh, condition, emb_w, emb_b);

    for (int l = 0; l < LREP; ++l) {
        size_t wofs = (size_t)l * HIDN * HIDN;
        k_ln<<<MTOK / 8, 256, 0, stream>>>(fx, ln1_g + l * HIDN, ln1_b + l * HIDN, h_bf);
        // xm = h @ px_w + px_b   (bf16 out)
        k_gemm<256, false, false, false, true><<<gGemm, 128, 0, stream>>>(
            h_bf, wpx_bf + wofs, px_b + l * HIDN, fx, xm_bf);
        // temperature + gumbel softmax routing
        k_perhead<<<(BB * HH * NN) / 256, 256, 0, stream>>>(
            xm_bf, gumbel, pt_w1, pt_b1, pt_w2, pt_b2, attn_bias, ps_w, ps_b, sw_bf, l);
        // st partials (WMMA K-reduction, deterministic chunking)
        k_slice_part<<<dim3(BB * HH, NCHUNK), 128, 0, stream>>>(sw_bf, xm_bf, st_part, norm_prt);
        // tiny attention over slice tokens (writes o transposed)
        k_attn<<<BB * HH, 32, 0, stream>>>(st_part, norm_prt, q_w, k_w, v_w, oT_bf, l);
        // scatter-back (WMMA, K=32) -> mid_bf used as ox
        k_scatter<<<dim3(BB * HH, NN / 64), 128, 0, stream>>>(sw_bf, oT_bf, mid_bf);
        // fx += ox @ o_w + o_b
        k_gemm<256, false, true, true, false><<<gGemm, 128, 0, stream>>>(
            mid_bf, wo_bf + wofs, o_b + l * HIDN, fx, xm_bf);
        // MLP
        k_ln<<<MTOK / 8, 256, 0, stream>>>(fx, ln2_g + l * HIDN, ln2_b + l * HIDN, h_bf);
        k_gemm<256, true, false, false, true><<<gGemm, 128, 0, stream>>>(
            h_bf, wm1_bf + wofs, mlp_b1 + l * HIDN, fx, mid_bf);
        k_gemm<256, false, true, true, false><<<gGemm, 128, 0, stream>>>(
            mid_bf, wm2_bf + wofs, mlp_b2 + l * HIDN, fx, xm_bf);
    }

    // Final LN + head
    k_ln<<<MTOK / 8, 256, 0, stream>>>(fx, ln3_g, ln3_b, h_bf);
    k_head<<<(MTOK * OUTC) / 256, 256, 0, stream>>>(h_bf, head_w, head_b, (float*)d_out);
}
